// FourierKANLayer_68126771249616
// MI455X (gfx1250) — matrix operand order repaired and checked by reading, not compiled, not run on hardware
//
#include <hip/hip_runtime.h>

typedef __attribute__((ext_vector_type(16))) _Float16 v16h;
typedef __attribute__((ext_vector_type(8)))  float    v8f;

#define N_ROWS   16384
#define I_DIM    128
#define O_DIM    128
#define G_DIM    16
#define K_TOT    4096      // 2 * I * G
#define K_STEP   32        // per step: one input i, cos(g) in K 0-15, sin(g) in K 16-31
#define N_STEPS  128       // = I_DIM (one K-step per input feature)
#define ROWS_PER_BLOCK 128
#define THREADS  256       // 8 wave32 waves

// ---------------------------------------------------------------------------
// Pack fp32 amplitudes (O,I,G) into f16 WMMA B-fragments with interleaved
// cos/sin K ordering: step ks covers input i = ks; within the step,
// k_local 0..15 = cos amplitudes (g = k_local), 16..31 = sin amplitudes.
// Fragment layout (ISA 16-bit B 32x16): lane l holds column N = l&15,
// halves e<8 hold K = 8*(l>>4) + e, e>=8 hold K = 16 + 8*(l>>4) + e-8 -> e<8 = cos, e>=8 = sin.
// ---------------------------------------------------------------------------
__global__ void fkan_pack_b(const float* __restrict__ cosA,
                            const float* __restrict__ sinA,
                            _Float16* __restrict__ Bp)
{
    int idx  = blockIdx.x * blockDim.x + threadIdx.x;   // one f16 element
    int e    =  idx        & 15;
    int lane = (idx >> 4)  & 31;
    int ot   = (idx >> 9)  & 7;
    int ks   =  idx >> 12;                              // = input index i
    int o  = ot * 16 + (lane & 15);
    int kl = (e < 8) ? ((lane >> 4) << 3) + e : 16 + ((lane >> 4) << 3) + (e - 8);   // 0..31
    int g  = kl & 15;                                   // harmonic index
    float v = (kl < 16) ? cosA[o * 2048 + ks * 16 + g]
                        : sinA[o * 2048 + ks * 16 + g];
    Bp[idx] = (_Float16)v;
}

// ---------------------------------------------------------------------------
// gfx1250 async global->LDS copy: each lane moves 2 x 16B. The instruction
// offset is added to BOTH the LDS and global addresses (ISA 10.x async ops),
// so one (lds, gaddr) pair covers both chunks. Tracked by ASYNCcnt.
// ---------------------------------------------------------------------------
__device__ __forceinline__ void async_copy_32B(unsigned lds_off,
                                               const _Float16* gsrc)
{
    asm volatile(
        "global_load_async_to_lds_b128 %0, %1, off\n\t"
        "global_load_async_to_lds_b128 %0, %1, off offset:16"
        :
        : "v"(lds_off), "v"(gsrc)
        : "memory");
}

__device__ __forceinline__ void wait_async0()
{
    asm volatile("s_wait_asynccnt 0x0" ::: "memory");
}

// ---------------------------------------------------------------------------
// Fused trig-feature GEMM + bias + LayerNorm.
// Block = 128 rows; each wave computes a 16x128 output tile with 8 WMMA accs.
// B tiles (8KB / K-step) are double-buffered through LDS via async loads, and
// B fragments are double-buffered in registers so a ds_load pair is always in
// flight underneath the current WMMA (waits settle at dscnt<=2, not 0).
// Per step, each lane computes ONE recurrence producing both the cos and sin
// harmonics it needs: A-layout lane-half h, element e<8 -> cos((8h+e+1)*th),
// e>=8 -> sin((8h+(e-8)+1)*th), matching the packed-B K ordering.
// ---------------------------------------------------------------------------
__global__ __launch_bounds__(THREADS) void fkan_gemm(
    const float* __restrict__ x, const _Float16* __restrict__ Bp,
    const float* __restrict__ bias, const float* __restrict__ gamma,
    const float* __restrict__ beta, float* __restrict__ out)
{
    __shared__ __align__(16) _Float16 Bs[2][THREADS * 16];   // 2 x 8 KB

    const int tid  = threadIdx.x;
    const int wave = tid >> 5;
    const int lane = tid & 31;
    const int half = lane >> 4;           // 0: lanes 0-15, 1: lanes 16-31
    const int lrow = lane & 15;
    const int n0   = blockIdx.x * ROWS_PER_BLOCK + wave * 16;
    const int row  = n0 + lrow;           // this lane's M row for the A fragment
    const float fbase = (float)(half * 8 + 1);   // starting harmonic (g+1)
    const float* __restrict__ xrow = x + (size_t)row * I_DIM;

    // LDS byte offsets of this thread's 32B staging slot in each buffer
    // (generic LDS pointer truncates to the LDS byte address, ISA 10.2).
    const unsigned bsOff0 = (unsigned)(uintptr_t)&Bs[0][0] + (unsigned)tid * 32u;
    const unsigned bsOff1 = bsOff0 + (unsigned)(THREADS * 16 * sizeof(_Float16));

    v8f acc[8] = {};

    // ---- prologue: async-stage K-step 0 of B into LDS buffer 0 ----------
    async_copy_32B(bsOff0, Bp + tid * 16);
    wait_async0();
    __syncthreads();

    for (int ks = 0; ks < N_STEPS; ++ks) {
        const int  cur      = ks & 1;
        const bool havenext = (ks + 1) < N_STEPS;

        // kick off the async copy of the next B tile into the other buffer;
        // the barrier at the end of the previous step guarantees no wave is
        // still reading it.
        if (havenext)
            async_copy_32B(cur ? bsOff0 : bsOff1,
                           Bp + (size_t)(ks + 1) * 4096 + tid * 16);

        const _Float16* __restrict__ bs = &Bs[cur][(unsigned)lane * 16];

        // prefetch first B fragment; its LDS latency hides under the trig chain
        v16h bf = *(const v16h*)&bs[0];

        // ---- build A fragment (16x32 f16): cos AND sin harmonics of one x
        const float th = xrow[ks];
        const float c1 = __cosf(th);
        const float s1 = __sinf(th);
        float cg = __cosf(th * fbase);     // cos((g0+1)*th)
        float sg = __sinf(th * fbase);

        v16h a;
#pragma unroll
        for (int j = 0; j < 8; ++j) {      // harmonics via angle addition
            a[j]     = (_Float16)cg;       // cos part -> K_local  8h+j
            a[8 + j] = (_Float16)sg;       // sin part -> K_local 16+8h+j
            const float cn = __builtin_fmaf(cg, c1, -sg * s1);
            const float sn = __builtin_fmaf(sg, c1,  cg * s1);
            cg = cn; sg = sn;
        }

        // ---- 8 WMMAs across O, with register-double-buffered B fragments
#pragma unroll
        for (int ot = 0; ot < 8; ++ot) {
            v16h bnext;
            if (ot < 7)
                bnext = *(const v16h*)&bs[(unsigned)(ot + 1) * 32 * 16];
            acc[ot] = __builtin_amdgcn_wmma_f32_16x16x32_f16(
                false, a, false, bf, (short)0, acc[ot], false, false);
            bf = bnext;
        }

        if (havenext) {
            wait_async0();        // this wave's async writes have landed in LDS
            __syncthreads();      // publish next tile to all waves
        }
    }

    // ---- epilogue: bias + LayerNorm over O, then store ------------------
    float bcol[8], gcol[8], zcol[8];
#pragma unroll
    for (int ot = 0; ot < 8; ++ot) {
        const int col = ot * 16 + lrow;
        bcol[ot] = bias[col]; gcol[ot] = gamma[col]; zcol[ot] = beta[col];
    }
#pragma unroll
    for (int ot = 0; ot < 8; ++ot)
#pragma unroll
        for (int r = 0; r < 8; ++r)
            acc[ot][r] += bcol[ot];

    // C/D layout: element r of acc = row (r + 8*half), column ot*16 + lrow.
    // Each output row lives entirely within one 16-lane half of the wave.
#pragma unroll
    for (int r = 0; r < 8; ++r) {
        float s = 0.f, s2 = 0.f;
#pragma unroll
        for (int ot = 0; ot < 8; ++ot) {
            const float v = acc[ot][r];
            s += v; s2 += v * v;
        }
#pragma unroll
        for (int m = 8; m >= 1; m >>= 1) {     // reduce within the 16-lane half
            s  += __shfl_xor(s,  m, 32);
            s2 += __shfl_xor(s2, m, 32);
        }
        const float mu   = s  * (1.0f / O_DIM);
        const float var  = s2 * (1.0f / O_DIM) - mu * mu;
        const float rstd = rsqrtf(var + 1e-5f);
        const int   n    = n0 + r + half * 8;
#pragma unroll
        for (int ot = 0; ot < 8; ++ot) {
            const int col = ot * 16 + lrow;
            out[(size_t)n * O_DIM + col] =
                (acc[ot][r] - mu) * rstd * gcol[ot] + zcol[ot];
        }
    }
}

// ---------------------------------------------------------------------------
extern "C" void kernel_launch(void* const* d_in, const int* in_sizes, int n_in,
                              void* d_out, int out_size, void* d_ws, size_t ws_size,
                              hipStream_t stream) {
    const float* x    = (const float*)d_in[0];
    const float* cosA = (const float*)d_in[1];
    const float* sinA = (const float*)d_in[2];
    const float* bias = (const float*)d_in[3];
    const float* gam  = (const float*)d_in[4];
    const float* bet  = (const float*)d_in[5];
    float*    out = (float*)d_out;
    _Float16* Bp  = (_Float16*)d_ws;   // 1 MB packed f16 B in workspace

    fkan_pack_b<<<(K_TOT * O_DIM) / 256, 256, 0, stream>>>(cosA, sinA, Bp);
    fkan_gemm<<<N_ROWS / ROWS_PER_BLOCK, THREADS, 0, stream>>>(
        x, Bp, bias, gam, bet, out);
}
